// GCN_56513179681533
// MI455X (gfx1250) — matrix operand order repaired and checked by reading, not compiled, not run on hardware
//
#include <hip/hip_runtime.h>
#include <stdint.h>

typedef __attribute__((ext_vector_type(16))) _Float16 v16h;
typedef __attribute__((ext_vector_type(8)))  float    v8f;

// ------------------------------------------------------------------
// Pack W (f32, row-major [K x Ncols]) into CDNA5 WMMA f16 B-fragments.
// Fragment layout per (coltile c, ktile t): 8 dwords x 32 lanes.
// Lane 0-15 : columns c*16+lane, K rows t*32 + {0..7, 16..23}  (dword r = {K=2r, K=2r+1}, +8 for r>=4)
// Lane16-31 : columns c*16+(lane-16), K rows t*32 + {8..15, 24..31}
// Stored at Bp[((c*4 + t)*8 + r)*32 + lane]  (per-r 128B coalesced).
// ------------------------------------------------------------------
__device__ inline uint32_t pack2_f16(float a, float b) {
    union { _Float16 h[2]; uint32_t u; } p;
    p.h[0] = (_Float16)a;
    p.h[1] = (_Float16)b;
    return p.u;
}

__global__ void prepack_kernel(const float* __restrict__ W1, const float* __restrict__ W2,
                               uint32_t* __restrict__ Bp1, uint32_t* __restrict__ Bp2) {
    int tid = blockIdx.x * blockDim.x + threadIdx.x;
    const float* W; uint32_t* Bp; int idx, ncols;
    if (tid < 8192)        { W = W1; Bp = Bp1; idx = tid;        ncols = 128; } // 8 ct * 4 kt * 8 r * 32 ln
    else if (tid < 12288)  { W = W2; Bp = Bp2; idx = tid - 8192; ncols = 64;  } // 4 ct
    else return;
    int lane = idx & 31;
    int r    = (idx >> 5) & 7;
    int t    = (idx >> 8) & 3;
    int c    = idx >> 10;
    int k    = t * 32 + ((lane >= 16) ? 8 : 0) + ((r >= 4) ? 8 : 0) + 2 * r;
    int col  = c * 16 + (lane & 15);
    Bp[idx] = pack2_f16(W[k * ncols + col], W[(k + 1) * ncols + col]);
}

// buf[i] = bias[i & colmask]  (seeds aggregation buffers with the bias; also
// re-initializes the poisoned output buffer deterministically every call)
__global__ void init_bias_kernel(float* __restrict__ buf, const float* __restrict__ bias,
                                 int total, int colmask) {
    int tid = blockIdx.x * blockDim.x + threadIdx.x;
    if (tid < total) buf[tid] = bias[tid & colmask];
}

// ------------------------------------------------------------------
// Out[ntiles*16 x (NCT*16)] = (RELU ? relu(A) : A) @ Wpacked, A f32 [rows x 128].
// Wave owns ONE column tile: its 4 B fragments (32 VGPRs) are loaded once and
// stay register-resident; row tiles of A stream past via a grid-stride loop.
// All NCT waves of a block read the same A rows -> WGP$ hits after first touch.
// ------------------------------------------------------------------
template<int NCT, bool RELU>
__global__ void gemm_wmma_kernel(const float* __restrict__ A,
                                 const uint32_t* __restrict__ Bp,
                                 float* __restrict__ Out, int ntiles) {
    constexpr int TPI = 8 / NCT;                // row tiles per block iteration
    const int wib  = threadIdx.x >> 5;          // wave in block: 0..7
    const int lane = threadIdx.x & 31;
    const int ct   = wib % NCT;                 // column tile owned by this wave
    const int tsub = wib / NCT;                 // row-tile sub-slot (NCT=4 -> 0/1)

    // One-time B preload for this column tile (register resident).
    v16h bfrag[4];
    #pragma unroll
    for (int t = 0; t < 4; ++t) {
        union { uint32_t u[8]; v16h h; } bu;
        const uint32_t* bp = Bp + (size_t)((ct * 4 + t) * 8) * 32 + lane;
        #pragma unroll
        for (int r = 0; r < 8; ++r) bu.u[r] = bp[r * 32];
        bfrag[t] = bu.h;
    }

    const int khalf = (lane >> 4) << 3;         // lanes 16-31 take the +8 K-halves
    const int ncols = NCT * 16;

    for (int tile = blockIdx.x * TPI + tsub; tile < ntiles; tile += gridDim.x * TPI) {
        const int rowbase = tile * 16;
        const int row     = rowbase + (lane & 15);
        v8f acc = {};
        #pragma unroll
        for (int t = 0; t < 4; ++t) {
            // A fragment: lane holds K = t*32 + khalf + {0..7} and + {16..23}
            const float* ap = A + (size_t)row * 128 + t * 32 + khalf;
            float4 x0 = *(const float4*)(ap);
            float4 x1 = *(const float4*)(ap + 4);
            float4 x2 = *(const float4*)(ap + 16);
            float4 x3 = *(const float4*)(ap + 20);
            float va[16] = { x0.x, x0.y, x0.z, x0.w,  x1.x, x1.y, x1.z, x1.w,
                             x2.x, x2.y, x2.z, x2.w,  x3.x, x3.y, x3.z, x3.w };
            v16h a;
            #pragma unroll
            for (int i = 0; i < 16; ++i) {
                float v = va[i];
                if (RELU) v = fmaxf(v, 0.0f);
                a[i] = (_Float16)v;
            }
            acc = __builtin_amdgcn_wmma_f32_16x16x32_f16(
                false, a, false, bfrag[t], (short)0, acc, false, false);
        }
        // C/D layout: dword r -> row rowbase+r (lanes 0-15) / rowbase+8+r (lanes 16-31)
        const int orow = rowbase + ((lane >> 4) << 3);
        const int col  = lane & 15;
        #pragma unroll
        for (int r = 0; r < 8; ++r)
            Out[(size_t)(orow + r) * ncols + ct * 16 + col] = acc[r];
    }
}

// ------------------------------------------------------------------
// F=128: one wave per edge, float4 gather, 4 f32 atomics per lane.
// ------------------------------------------------------------------
__global__ void edge_scatter128_kernel(const long long* __restrict__ ei,
                                       const float* __restrict__ ew,
                                       const float* __restrict__ sup,
                                       float* __restrict__ agg, int E) {
    int wid  = (blockIdx.x * blockDim.x + threadIdx.x) >> 5;
    int lane = threadIdx.x & 31;
    if (wid >= E) return;
    long long dst = ei[wid];                    // edge_index[0][e]
    long long src = ei[(long long)E + wid];     // edge_index[1][e]
    float     w   = ew[wid];
    const float4 s = *(const float4*)(sup + src * 128 + lane * 4);
    float* ap = agg + dst * 128 + lane * 4;
    atomicAdd(ap + 0, w * s.x);
    atomicAdd(ap + 1, w * s.y);
    atomicAdd(ap + 2, w * s.z);
    atomicAdd(ap + 3, w * s.w);
}

// ------------------------------------------------------------------
// F=64: two edges per wave (each 16-lane half owns one edge), float4 gather.
// ------------------------------------------------------------------
__global__ void edge_scatter64_kernel(const long long* __restrict__ ei,
                                      const float* __restrict__ ew,
                                      const float* __restrict__ sup,
                                      float* __restrict__ agg, int E) {
    int wid  = (blockIdx.x * blockDim.x + threadIdx.x) >> 5;
    int lane = threadIdx.x & 31;
    int e    = wid * 2 + (lane >> 4);
    if (e >= E) return;
    long long dst = ei[e];
    long long src = ei[(long long)E + e];
    float     w   = ew[e];
    int f = (lane & 15) * 4;
    const float4 s = *(const float4*)(sup + src * 64 + f);
    float* ap = agg + dst * 64 + f;
    atomicAdd(ap + 0, w * s.x);
    atomicAdd(ap + 1, w * s.y);
    atomicAdd(ap + 2, w * s.z);
    atomicAdd(ap + 3, w * s.w);
}

extern "C" void kernel_launch(void* const* d_in, const int* in_sizes, int n_in,
                              void* d_out, int out_size, void* d_ws, size_t ws_size,
                              hipStream_t stream) {
    (void)n_in; (void)out_size; (void)ws_size;
    const float*     x  = (const float*)d_in[0];      // [N,128] f32
    const long long* ei = (const long long*)d_in[1];  // [2,E] int64
    const float*     ew = (const float*)d_in[2];      // [E] f32
    const float*     W1 = (const float*)d_in[3];      // [128,128]
    const float*     b1 = (const float*)d_in[4];      // [128]
    const float*     W2 = (const float*)d_in[5];      // [128,64]
    const float*     b2 = (const float*)d_in[6];      // [64]
    float*           out = (float*)d_out;             // [N,64]

    const int N = in_sizes[0] / 128;
    const int E = in_sizes[2];
    const int ntiles = N / 16;                        // N = 100000 -> 6250 exact

    // Workspace layout (~128 MB): support1 | agg1 | support2 | Bp1 | Bp2
    float*    support1 = (float*)d_ws;
    float*    agg1     = support1 + (size_t)N * 128;
    float*    support2 = agg1     + (size_t)N * 128;
    uint32_t* Bp1      = (uint32_t*)(support2 + (size_t)N * 64);
    uint32_t* Bp2      = Bp1 + 8192;

    // Grid sizing: enough blocks to fill the device, grid-stride over row tiles
    // (amortizes the register-resident B preload).
    int g1 = ntiles        < 1280 ? ntiles            : 1280;   // NCT=8: 1 tile/iter
    int g2 = (ntiles + 1)/2 < 1280 ? (ntiles + 1) / 2 : 1280;   // NCT=4: 2 tiles/iter
    const int eb128 = (E + 7) / 8;                    // wave per edge
    const int eb64  = (E + 15) / 16;                  // 2 edges per wave

    prepack_kernel<<<48, 256, 0, stream>>>(W1, W2, Bp1, Bp2);
    init_bias_kernel<<<(N * 128 + 255) / 256, 256, 0, stream>>>(agg1, b1, N * 128, 127);
    init_bias_kernel<<<(N * 64 + 255) / 256, 256, 0, stream>>>(out, b2, N * 64, 63);

    // Layer 1: support1 = x @ W1 ; agg1 (pre-seeded with b1) += scatter(ew * support1[src])
    gemm_wmma_kernel<8, false><<<g1, 256, 0, stream>>>(x, Bp1, support1, ntiles);
    edge_scatter128_kernel<<<eb128, 256, 0, stream>>>(ei, ew, support1, agg1, E);

    // Layer 2: support2 = relu(agg1) @ W2 ; out (pre-seeded with b2) += scatter(...)
    gemm_wmma_kernel<4, true><<<g2, 256, 0, stream>>>(agg1, Bp2, support2, ntiles);
    edge_scatter64_kernel<<<eb64, 256, 0, stream>>>(ei, ew, support2, out, E);
}
